// GATEncoder_13743895347751
// MI455X (gfx1250) — matrix operand order repaired and checked by reading, not compiled, not run on hardware
//
#include <hip/hip_runtime.h>
#include <cstdint>

#define TPB 256

typedef __attribute__((ext_vector_type(16))) __bf16 v16bf;
typedef __attribute__((ext_vector_type(8)))  float  v8f;

union FragB32 { uint4 q[2]; v16bf v; };

// ---- monotonic float<->uint encoding for atomicMax-based segment max ----
__device__ __forceinline__ unsigned f32_ord(float f) {
  unsigned b = __float_as_uint(f);
  return (b & 0x80000000u) ? ~b : (b | 0x80000000u);
}
__device__ __forceinline__ float ord_f32(unsigned u) {
  unsigned b = (u & 0x80000000u) ? (u & 0x7FFFFFFFu) : ~u;
  return __uint_as_float(b);
}

// ---------------- utility kernels ----------------
__global__ void zero_f32(float* __restrict__ p, int n) {
  int i = blockIdx.x * TPB + threadIdx.x;
  if (i < n) p[i] = 0.0f;
}

__global__ void f32_to_bf16_k(const float* __restrict__ in, __bf16* __restrict__ out, int n) {
  int i = blockIdx.x * TPB + threadIdx.x;
  if (i < n) out[i] = (__bf16)in[i];
}

// Pack W (row-major K x NCOLS, f32) into WMMA B-fragment layout (bf16 pairs).
// Bpack[((kt*NT + nt)*32 + lane)*8 + d] holds K = kt*32 + (lane>>4)*16 + 2d, 2d+1
// for column n = nt*16 + (lane&15).
__global__ void pack_w_bf16(const float* __restrict__ W, uint32_t* __restrict__ out,
                            int K, int NCOLS) {
  int NT = NCOLS / 16;
  int total = (K / 32) * NT * 256;
  int tid = blockIdx.x * TPB + threadIdx.x;
  if (tid >= total) return;
  int d  = tid & 7;
  int l  = (tid >> 3) & 31;
  int nt = (tid >> 8) % NT;
  int kt = tid / (256 * NT);
  int n    = nt * 16 + (l & 15);
  int half = l >> 4;
  int k    = kt * 32 + half * 16 + 2 * d;
  union { __bf16 h[2]; uint32_t u; } p;
  p.h[0] = (__bf16)W[(size_t)k * NCOLS + n];
  p.h[1] = (__bf16)W[(size_t)(k + 1) * NCOLS + n];
  out[tid] = p.u;
}

// ---------------- WMMA GEMM: C[M,NCOLS] = A[M,KDIM](bf16) * Bpack ----------------
// 8 waves per block; wave w computes rows [tile*16, tile*16+16) x all NCOLS.
template<int KDIM, int NCOLS>
__global__ __launch_bounds__(TPB) void gemm_bf16_wmma(const __bf16* __restrict__ A,
                                                      const uint32_t* __restrict__ Bpack,
                                                      float* __restrict__ C, int mtiles) {
  constexpr int NT = NCOLS / 16;
  constexpr int KT = KDIM / 32;
  int wave = threadIdx.x >> 5;
  int lane = threadIdx.x & 31;
  int tile = blockIdx.x * 8 + wave;
  if (tile >= mtiles) return;
  int m    = lane & 15;
  int half = lane >> 4;
  const __bf16* arow = A + (size_t)(tile * 16 + m) * KDIM;

  v8f acc[NT] = {};
  for (int kt = 0; kt < KT; ++kt) {
    FragB32 af;
    af.q[0] = *(const uint4*)(arow + kt * 32 + half * 16);       // K = kt*32 + half*16 .. +7
    af.q[1] = *(const uint4*)(arow + kt * 32 + half * 16 + 8);   // K = kt*32+half*16+8 .. +15
#pragma unroll
    for (int nt = 0; nt < NT; ++nt) {
      FragB32 bf;
      const uint4* bp = (const uint4*)(Bpack + ((size_t)(kt * NT + nt) * 32 + lane) * 8);
      bf.q[0] = bp[0];
      bf.q[1] = bp[1];
      acc[nt] = __builtin_amdgcn_wmma_f32_16x16x32_bf16(
          false, af.v, false, bf.v, (short)0, acc[nt], false, false);
    }
  }
#pragma unroll
  for (int nt = 0; nt < NT; ++nt)
#pragma unroll
    for (int v = 0; v < 8; ++v)
      C[(size_t)(tile * 16 + v + half * 8) * NCOLS + nt * 16 + m] = acc[nt][v];
}

// ---------------- attention scalars: als/ald[n*H+h] = dot(h[n,h,:], a_{src,dst}[h,:]) ----
__global__ void node_att(const float* __restrict__ h, const float* __restrict__ a_src,
                         const float* __restrict__ a_dst, float* __restrict__ als,
                         float* __restrict__ ald, int Nn, int H, int C) {
  int gid = blockIdx.x * TPB + threadIdx.x;
  if (gid >= Nn * H) return;
  int n = gid / H, hh = gid - n * H;
  const float* row = h + (size_t)n * H * C + (size_t)hh * C;
  const float* as  = a_src + hh * C;
  const float* ad  = a_dst + hh * C;
  float ss = 0.f, sd = 0.f;
  for (int c = 0; c < C; ++c) {
    float v = row[c];
    ss += v * as[c];
    sd += v * ad[c];
  }
  als[gid] = ss;
  ald[gid] = sd;
}

// ---------------- edge pass A: leaky-relu logits + segment max ----------------
__global__ void edge_logits(const int* __restrict__ src, const int* __restrict__ dst,
                            const float* __restrict__ als, const float* __restrict__ ald,
                            float* __restrict__ ebuf, unsigned* __restrict__ menc,
                            int E, int Et, int H) {
  int gid = blockIdx.x * TPB + threadIdx.x;
  if (gid >= Et * H) return;
  int eid = gid / H, hh = gid - eid * H;
  int s, d;
  if (eid < E) { s = src[eid]; d = dst[eid]; } else { s = d = eid - E; }
  float e = als[s * H + hh] + ald[d * H + hh];
  e = e > 0.f ? e : 0.2f * e;
  ebuf[gid] = e;
  atomicMax(&menc[d * H + hh], f32_ord(e));
}

// ---------------- edge pass B: exp(e - m) + segment sum of weights ----------------
__global__ void edge_exp(const int* __restrict__ src, const int* __restrict__ dst,
                         float* __restrict__ ebuf, const unsigned* __restrict__ menc,
                         float* __restrict__ z, int E, int Et, int H) {
  int gid = blockIdx.x * TPB + threadIdx.x;
  if (gid >= Et * H) return;
  int eid = gid / H, hh = gid - eid * H;
  int d = (eid < E) ? dst[eid] : eid - E;
  float m = ord_f32(menc[d * H + hh]);
  float w = __expf(ebuf[gid] - m);
  ebuf[gid] = w;
  atomicAdd(&z[d * H + hh], w);
}

// ---------------- edge pass C: scatter unnormalized messages ----------------
__global__ void edge_scatter(const int* __restrict__ src, const int* __restrict__ dst,
                             const float* __restrict__ w, const float* __restrict__ feat,
                             float* __restrict__ acc, int E, int Et, int H, int Ctot) {
  int gid = blockIdx.x * TPB + threadIdx.x;
  if (gid >= Et * Ctot) return;
  int eid = gid / Ctot;
  int c = gid - eid * Ctot;
  int s, d;
  if (eid < E) { s = src[eid]; d = dst[eid]; } else { s = d = eid - E; }
  int head = c / (Ctot / H);
  float ww = w[eid * H + head];
  atomicAdd(&acc[(size_t)d * Ctot + c], ww * feat[(size_t)s * Ctot + c]);
}

// ---------------- layer-1 finalize: /z + bias, ELU, write bf16 ----------------
__global__ void finalize_elu_bf16(const float* __restrict__ acc, const float* __restrict__ z,
                                  const float* __restrict__ bias, __bf16* __restrict__ out_bf,
                                  int Nn, int H, int C) {
  int gid = blockIdx.x * TPB + threadIdx.x;
  int Ctot = H * C;
  if (gid >= Nn * Ctot) return;
  int n = gid / Ctot;
  int c = gid - n * Ctot;
  int head = c / C;
  float v = acc[gid] / z[n * H + head] + bias[c];
  v = v > 0.f ? v : expm1f(v);
  out_bf[gid] = (__bf16)v;
}

// ---------------- layer-2 finalize: /z + bias ----------------
__global__ void finalize_out(const float* __restrict__ acc, const float* __restrict__ z,
                             const float* __restrict__ bias, float* __restrict__ out,
                             int Nn, int C) {
  int gid = blockIdx.x * TPB + threadIdx.x;
  if (gid >= Nn * C) return;
  int n = gid / C;
  int c = gid - n * C;
  out[gid] = acc[gid] / z[n] + bias[c];
}

static inline unsigned gdim(long long n) { return (unsigned)((n + TPB - 1) / TPB); }

extern "C" void kernel_launch(void* const* d_in, const int* in_sizes, int n_in,
                              void* d_out, int out_size, void* d_ws, size_t ws_size,
                              hipStream_t stream) {
  const float* x      = (const float*)d_in[0];
  const int*   ei     = (const int*)d_in[1];
  const float* W1     = (const float*)d_in[2];
  const float* a_src1 = (const float*)d_in[3];
  const float* a_dst1 = (const float*)d_in[4];
  const float* b1     = (const float*)d_in[5];
  const float* W2     = (const float*)d_in[6];
  const float* a_src2 = (const float*)d_in[7];
  const float* a_dst2 = (const float*)d_in[8];
  const float* b2     = (const float*)d_in[9];
  float* out = (float*)d_out;

  const int Nn  = 100000;
  const int E   = in_sizes[1] / 2;   // edge_index is [2,E]
  const int Et  = E + Nn;            // with self-loops
  const int DIN = 256, C1 = 128, H1h = 2, C2 = 64;
  const int MT  = Nn / 16;           // 6250 M-tiles (exact)

  const int* src = ei;
  const int* dst = ei + E;

  // ---------------- workspace layout ----------------
  char* ws = (char*)d_ws;
  size_t off = 0;
  float* h1 = (float*)(ws + off);           off += (size_t)Nn * C1 * 4;   // 51.2 MB
  float* acc1 = (float*)(ws + off);         off += (size_t)Nn * C1 * 4;   // 51.2 MB (reused: h2, acc2)
  __bf16* xbf = (__bf16*)(ws + off);        off += (size_t)Nn * DIN * 2;  // 51.2 MB (reused: h1elu bf16)
  float* al_s1 = (float*)(ws + off);        off += (size_t)2 * Nn * 4;
  float* al_d1 = (float*)(ws + off);        off += (size_t)2 * Nn * 4;
  unsigned* m1 = (unsigned*)(ws + off);     off += (size_t)2 * Nn * 4;
  float* z1 = (float*)(ws + off);           off += (size_t)2 * Nn * 4;
  float* al_s2 = (float*)(ws + off);        off += (size_t)Nn * 4;
  float* al_d2 = (float*)(ws + off);        off += (size_t)Nn * 4;
  unsigned* m2 = (unsigned*)(ws + off);     off += (size_t)Nn * 4;
  float* z2 = (float*)(ws + off);           off += (size_t)Nn * 4;
  float* e1 = (float*)(ws + off);           off += (size_t)Et * 2 * 4;    // 13.6 MB (reused: e2)
  uint32_t* w1pack = (uint32_t*)(ws + off); off += (size_t)DIN * C1 / 2 * 4;
  uint32_t* w2pack = (uint32_t*)(ws + off); off += (size_t)C1 * C2 / 2 * 4;

  __bf16* h1bf = xbf;                 // reuse: layer-2 GEMM input (written after GEMM1 done)
  float* h2   = acc1;                 // reuse: [Nn, 64]
  float* acc2 = acc1 + (size_t)Nn * C2;
  float* e2   = e1;

  // ---------------- layer 1 ----------------
  f32_to_bf16_k<<<gdim((long long)Nn * DIN), TPB, 0, stream>>>(x, xbf, Nn * DIN);
  pack_w_bf16<<<gdim(DIN * C1 / 2), TPB, 0, stream>>>(W1, w1pack, DIN, C1);
  pack_w_bf16<<<gdim(C1 * C2 / 2), TPB, 0, stream>>>(W2, w2pack, C1, C2);

  gemm_bf16_wmma<256, 128><<<(MT + 7) / 8, TPB, 0, stream>>>(xbf, w1pack, h1, MT);

  node_att<<<gdim((long long)Nn * H1h), TPB, 0, stream>>>(h1, a_src1, a_dst1,
                                                          al_s1, al_d1, Nn, H1h, 64);

  zero_f32<<<gdim((long long)Nn * C1), TPB, 0, stream>>>(acc1, Nn * C1);
  zero_f32<<<gdim((long long)4 * Nn), TPB, 0, stream>>>((float*)m1, 4 * Nn); // m1 + z1

  edge_logits<<<gdim((long long)Et * H1h), TPB, 0, stream>>>(src, dst, al_s1, al_d1,
                                                             e1, m1, E, Et, H1h);
  edge_exp<<<gdim((long long)Et * H1h), TPB, 0, stream>>>(src, dst, e1, m1, z1, E, Et, H1h);
  edge_scatter<<<gdim((long long)Et * C1), TPB, 0, stream>>>(src, dst, e1, h1, acc1,
                                                             E, Et, H1h, C1);
  finalize_elu_bf16<<<gdim((long long)Nn * C1), TPB, 0, stream>>>(acc1, z1, b1, h1bf,
                                                                  Nn, H1h, 64);

  // ---------------- layer 2 ----------------
  gemm_bf16_wmma<128, 64><<<(MT + 7) / 8, TPB, 0, stream>>>(h1bf, w2pack, h2, MT);

  node_att<<<gdim((long long)Nn), TPB, 0, stream>>>(h2, a_src2, a_dst2,
                                                    al_s2, al_d2, Nn, 1, 64);

  zero_f32<<<gdim((long long)Nn * C2), TPB, 0, stream>>>(acc2, Nn * C2);
  zero_f32<<<gdim((long long)2 * Nn), TPB, 0, stream>>>((float*)m2, 2 * Nn); // m2 + z2

  edge_logits<<<gdim((long long)Et), TPB, 0, stream>>>(src, dst, al_s2, al_d2,
                                                       e2, m2, E, Et, 1);
  edge_exp<<<gdim((long long)Et), TPB, 0, stream>>>(src, dst, e2, m2, z2, E, Et, 1);
  edge_scatter<<<gdim((long long)Et * C2), TPB, 0, stream>>>(src, dst, e2, h2, acc2,
                                                             E, Et, 1, C2);
  finalize_out<<<gdim((long long)Nn * C2), TPB, 0, stream>>>(acc2, z2, b2, out, Nn, C2);
}
